// SiameseLSTM_58677843198486
// MI455X (gfx1250) — matrix operand order repaired and checked by reading, not compiled, not run on hardware
//
#include <hip/hip_runtime.h>
#include <hip/hip_bf16.h>

// ---------------------------------------------------------------------------
// Siamese LSTM for MI455X (gfx1250), wave32 + bf16 WMMA (16x16x32, f32 acc).
//
// Phases:
//   1) prep_weights : W_ih -> bf16 [2048][320] (K padded 300->320, zeros),
//                     W_hh -> bf16 [2048][512], bias = b_ih + b_hh (f32)
//   2) prep_state   : H_bf[buf0] <- bf16(h0_a|h0_b) rows 0..511, C <- f32 c0
//   3) prep_embed   : X[2][64][256][320] <- bf16(emb_table[s]) (pad zeros)
//   4) 64x lstm_step: gates = bias + X_t @ Wih^T + h_prev @ Whh^T  (WMMA,
//                     software-pipelined fragment loads), LSTM cell fused
//                     after LDS staging; h ping-pong bf16.
//   5) final_out    : out[b] = exp(-sum |ha - hb|)
// ---------------------------------------------------------------------------

typedef __bf16 bf16_t;
typedef bf16_t v16bf __attribute__((ext_vector_type(16)));
typedef float  v8f   __attribute__((ext_vector_type(8)));

#define T_STEPS 64
#define BATCH   256
#define EMB_D   300
#define HID     512
#define G4      2048      // 4*HID
#define KXP     320       // padded embedding K
#define MROWS   512       // 2 sequences * BATCH
#define NKX     (KXP / 32)   // 10 embedding k-chunks
#define NKH     (HID / 32)   // 16 recurrent k-chunks

// workspace byte offsets (all 16B aligned)
#define OFF_WIH  0u                    // 2048*320*2  = 1310720
#define OFF_WHH  1310720u              // 2048*512*2  = 2097152
#define OFF_BIAS 3407872u              // 2048*4      = 8192
#define OFF_X    3416064u              // 2*64*256*320*2 = 20971520
#define OFF_HB   24387584u             // 2*512*512*2 = 1048576 (ping-pong)
#define OFF_C    25436160u             // 512*512*4   = 1048576
#define OFF_HF   26484736u             // 512*512*4   = 1048576

// ---------------------------------------------------------------------------
__global__ void prep_weights(const float* __restrict__ Wih,
                             const float* __restrict__ Whh,
                             const float* __restrict__ bih,
                             const float* __restrict__ bhh,
                             bf16_t* __restrict__ WihB,
                             bf16_t* __restrict__ WhhB,
                             float* __restrict__ bias) {
    const int total = G4 * KXP + G4 * HID + G4;
    for (int i = blockIdx.x * blockDim.x + threadIdx.x; i < total;
         i += gridDim.x * blockDim.x) {
        if (i < G4 * KXP) {
            int n = i / KXP, k = i - n * KXP;
            WihB[i] = (bf16_t)((k < EMB_D) ? Wih[n * EMB_D + k] : 0.0f);
        } else if (i < G4 * KXP + G4 * HID) {
            int j = i - G4 * KXP;
            WhhB[j] = (bf16_t)Whh[j];
        } else {
            int n = i - G4 * KXP - G4 * HID;
            bias[n] = bih[n] + bhh[n];
        }
    }
}

__global__ void prep_state(const float* __restrict__ h0a,
                           const float* __restrict__ c0a,
                           const float* __restrict__ h0b,
                           const float* __restrict__ c0b,
                           bf16_t* __restrict__ Hb0,
                           float* __restrict__ C) {
    const int total = MROWS * HID;
    for (int i = blockIdx.x * blockDim.x + threadIdx.x; i < total;
         i += gridDim.x * blockDim.x) {
        int row = i >> 9;           // /512
        int k   = i & 511;
        float h, c;
        if (row < BATCH) { h = h0a[row * HID + k]; c = c0a[row * HID + k]; }
        else { int r = row - BATCH; h = h0b[r * HID + k]; c = c0b[r * HID + k]; }
        Hb0[i] = (bf16_t)h;
        C[i]   = c;
    }
}

__global__ void prep_embed(const int* __restrict__ s1,
                           const int* __restrict__ s2,
                           const float* __restrict__ emb,
                           bf16_t* __restrict__ X) {
    const long total = 2L * T_STEPS * BATCH * KXP;
    const long stride = (long)gridDim.x * blockDim.x;
    for (long i = blockIdx.x * (long)blockDim.x + threadIdx.x; i < total;
         i += stride) {
        int  k   = (int)(i % KXP);
        long row = i / KXP;
        int  b   = (int)(row & (BATCH - 1));
        int  t   = (int)((row >> 8) & (T_STEPS - 1));
        int  seq = (int)(row >> 14);                 // / (256*64)
        int  tok = (seq ? s2 : s1)[t * BATCH + b];
        X[i] = (bf16_t)((k < EMB_D) ? emb[(long)tok * EMB_D + k] : 0.0f);
    }
}

// ---------------------------------------------------------------------------
union F16 { v16bf v; uint4 q[2]; };

// A-fragment: two 16B chunks 32B apart (K {0..7|8..15} and {16..23|24..31})
__device__ __forceinline__ void load_a(F16& f, const bf16_t* p, int koff) {
    f.q[0] = *(const uint4*)(p + koff);
    f.q[1] = *(const uint4*)(p + koff + 16);
}
// B-fragment: two 16B chunks 32B apart (K {0..7|8..15} and {16..23|24..31})
__device__ __forceinline__ void load_b(F16& f, const bf16_t* p, int koff) {
    f.q[0] = *(const uint4*)(p + koff);
    f.q[1] = *(const uint4*)(p + koff + 16);
}

__device__ __forceinline__ float fast_sigmoid(float x) {
    return 1.0f / (1.0f + __expf(-x));
}
__device__ __forceinline__ float fast_tanh(float x) {
    return 1.0f - 2.0f / (__expf(2.0f * x) + 1.0f);
}

// ---------------------------------------------------------------------------
// One recurrent timestep for BOTH sequences, fused GEMM + LSTM cell.
// grid = (16 col-strips, 4 row-tiles), block = 256 threads (8 waves).
// WG tile: 128 rows x (4 gates x 32 cols). Wave tile: 32 rows x 64 cols
// = 2x4 WMMA 16x16 tiles. Fragment loads double-buffered one k-chunk ahead.
// ---------------------------------------------------------------------------
__launch_bounds__(256, 1)
__global__ void lstm_step(const bf16_t* __restrict__ WihB,
                          const bf16_t* __restrict__ WhhB,
                          const float*  __restrict__ bias,
                          const bf16_t* __restrict__ X,
                          bf16_t* __restrict__ Hb,
                          float*  __restrict__ C,
                          float*  __restrict__ Hf,
                          int t) {
    __shared__ float lds[128 * 128];   // gate tile, 64 KB

    const int tid  = threadIdx.x;
    const int lane = tid & 31;
    const int wave = tid >> 5;
    const int l16  = lane & 15;
    const int lhi  = lane >> 4;        // 0 | 1
    const int ns   = blockIdx.x;       // column strip 0..15
    const int wgM  = blockIdx.y;       // row tile 0..3

    const int mw    = wgM * 128 + (wave & 3) * 32;  // wave row base
    const int gpair = wave >> 2;                    // gates {0,1} or {2,3}

    // --- accumulators, initialized with bias (covers b_ih + b_hh) ---------
    v8f acc[2][4];
#pragma unroll
    for (int ct = 0; ct < 4; ++ct) {
        int g = gpair * 2 + (ct >> 1);
        int colbase = g * HID + ns * 32 + (ct & 1) * 16;
        float bv = bias[colbase + l16];
        v8f a;
#pragma unroll
        for (int j = 0; j < 8; ++j) a[j] = bv;
        acc[0][ct] = a;
        acc[1][ct] = a;
    }

    // --- A-side row pointers (two 16-row subtiles per wave) ---------------
    const bf16_t* Hprev = Hb + (size_t)(t & 1) * MROWS * HID;
    const bf16_t* xrow[2];
    const bf16_t* hrow[2];
#pragma unroll
    for (int i = 0; i < 2; ++i) {
        int row = mw + i * 16 + l16;
        int seq = row >> 8;
        int b   = row & 255;
        xrow[i] = X + (((size_t)seq * T_STEPS + t) * BATCH + b) * KXP;
        hrow[i] = Hprev + (size_t)row * HID;
    }

    // --- B-side column pointers (4 col subtiles) --------------------------
    const bf16_t* bcolX[4];
    const bf16_t* bcolH[4];
#pragma unroll
    for (int ct = 0; ct < 4; ++ct) {
        int g = gpair * 2 + (ct >> 1);
        int n = g * HID + ns * 32 + (ct & 1) * 16 + l16;
        bcolX[ct] = WihB + (size_t)n * KXP;
        bcolH[ct] = WhhB + (size_t)n * HID;
    }

    // --- software-pipelined K loop ----------------------------------------
    F16 aCur[2], bCur[4];
    {
        const int ka = lhi * 8, kb = lhi * 8;
#pragma unroll
        for (int i = 0; i < 2; ++i)  load_a(aCur[i], xrow[i], ka);
#pragma unroll
        for (int ct = 0; ct < 4; ++ct) load_b(bCur[ct], bcolX[ct], kb);
    }

    // K loop 1: embedding part, K = 320; last iter prefetches first H-chunk
#pragma unroll
    for (int kc = 0; kc < NKX; ++kc) {
        F16 aNxt[2], bNxt[4];
        if (kc + 1 < NKX) {
            const int ka = (kc + 1) * 32 + lhi * 8;
            const int kb = (kc + 1) * 32 + lhi * 8;
#pragma unroll
            for (int i = 0; i < 2; ++i)  load_a(aNxt[i], xrow[i], ka);
#pragma unroll
            for (int ct = 0; ct < 4; ++ct) load_b(bNxt[ct], bcolX[ct], kb);
        } else {                       // bridge into recurrent K loop
            const int ka = lhi * 8, kb = lhi * 8;
#pragma unroll
            for (int i = 0; i < 2; ++i)  load_a(aNxt[i], hrow[i], ka);
#pragma unroll
            for (int ct = 0; ct < 4; ++ct) load_b(bNxt[ct], bcolH[ct], kb);
        }
#pragma unroll
        for (int ct = 0; ct < 4; ++ct)
#pragma unroll
            for (int i = 0; i < 2; ++i)
                acc[i][ct] = __builtin_amdgcn_wmma_f32_16x16x32_bf16(
                    false, aCur[i].v, false, bCur[ct].v, (short)0, acc[i][ct],
                    false, false);
#pragma unroll
        for (int i = 0; i < 2; ++i)  aCur[i] = aNxt[i];
#pragma unroll
        for (int ct = 0; ct < 4; ++ct) bCur[ct] = bNxt[ct];
    }

    // K loop 2: recurrent part, K = 512 (chunk 0 already resident)
#pragma unroll
    for (int kc = 0; kc < NKH; ++kc) {
        F16 aNxt[2], bNxt[4];
        if (kc + 1 < NKH) {
            const int ka = (kc + 1) * 32 + lhi * 8;
            const int kb = (kc + 1) * 32 + lhi * 8;
#pragma unroll
            for (int i = 0; i < 2; ++i)  load_a(aNxt[i], hrow[i], ka);
#pragma unroll
            for (int ct = 0; ct < 4; ++ct) load_b(bNxt[ct], bcolH[ct], kb);
        }
#pragma unroll
        for (int ct = 0; ct < 4; ++ct)
#pragma unroll
            for (int i = 0; i < 2; ++i)
                acc[i][ct] = __builtin_amdgcn_wmma_f32_16x16x32_bf16(
                    false, aCur[i].v, false, bCur[ct].v, (short)0, acc[i][ct],
                    false, false);
        if (kc + 1 < NKH) {
#pragma unroll
            for (int i = 0; i < 2; ++i)  aCur[i] = aNxt[i];
#pragma unroll
            for (int ct = 0; ct < 4; ++ct) bCur[ct] = bNxt[ct];
        }
    }

    // --- stage gate tile to LDS -------------------------------------------
    // C/D layout: lane l, VGPR j -> M = j + (l/16)*8, N = l%16
#pragma unroll
    for (int i = 0; i < 2; ++i) {
        int rbase = (wave & 3) * 32 + i * 16 + lhi * 8;
#pragma unroll
        for (int ct = 0; ct < 4; ++ct) {
            int g    = gpair * 2 + (ct >> 1);
            int cidx = g * 32 + (ct & 1) * 16 + l16;
#pragma unroll
            for (int j = 0; j < 8; ++j)
                lds[(rbase + j) * 128 + cidx] = acc[i][ct][j];
        }
    }
    __syncthreads();

    // --- fused LSTM cell update -------------------------------------------
    bf16_t* Hnext = Hb + (size_t)((t + 1) & 1) * MROWS * HID;
    for (int e = tid; e < 128 * 32; e += 256) {
        int r = e >> 5;
        int c = e & 31;
        float gi = lds[r * 128 +      c];
        float gf = lds[r * 128 + 32 + c];
        float gg = lds[r * 128 + 64 + c];
        float go = lds[r * 128 + 96 + c];
        int grow = wgM * 128 + r;
        int gcol = ns * 32 + c;
        size_t sidx = (size_t)grow * HID + gcol;
        float cp = C[sidx];
        float cn = fast_sigmoid(gf) * cp + fast_sigmoid(gi) * fast_tanh(gg);
        float hn = fast_sigmoid(go) * fast_tanh(cn);
        C[sidx]     = cn;
        Hnext[sidx] = (bf16_t)hn;
        if (t == T_STEPS - 1) Hf[sidx] = hn;
    }
}

// ---------------------------------------------------------------------------
__global__ void final_out(const float* __restrict__ Hf, float* __restrict__ out) {
    __shared__ float red[256];
    const int b = blockIdx.x;
    float s = 0.0f;
    for (int j = threadIdx.x; j < HID; j += 256)
        s += fabsf(Hf[(size_t)b * HID + j] - Hf[(size_t)(b + BATCH) * HID + j]);
    red[threadIdx.x] = s;
    __syncthreads();
    for (int st = 128; st > 0; st >>= 1) {
        if (threadIdx.x < st) red[threadIdx.x] += red[threadIdx.x + st];
        __syncthreads();
    }
    if (threadIdx.x == 0) out[b] = __expf(-red[0]);
}

// ---------------------------------------------------------------------------
extern "C" void kernel_launch(void* const* d_in, const int* in_sizes, int n_in,
                              void* d_out, int out_size, void* d_ws,
                              size_t ws_size, hipStream_t stream) {
    const int*   s1  = (const int*)d_in[0];
    const int*   s2  = (const int*)d_in[1];
    const float* emb = (const float*)d_in[2];
    const float* Wih = (const float*)d_in[3];
    const float* Whh = (const float*)d_in[4];
    const float* bih = (const float*)d_in[5];
    const float* bhh = (const float*)d_in[6];
    const float* h0a = (const float*)d_in[7];
    const float* c0a = (const float*)d_in[8];
    const float* h0b = (const float*)d_in[9];
    const float* c0b = (const float*)d_in[10];
    float* out = (float*)d_out;

    char* ws = (char*)d_ws;
    bf16_t* WihB = (bf16_t*)(ws + OFF_WIH);
    bf16_t* WhhB = (bf16_t*)(ws + OFF_WHH);
    float*  bias = (float*)(ws + OFF_BIAS);
    bf16_t* X    = (bf16_t*)(ws + OFF_X);
    bf16_t* Hb   = (bf16_t*)(ws + OFF_HB);
    float*  C    = (float*)(ws + OFF_C);
    float*  Hf   = (float*)(ws + OFF_HF);

    prep_weights<<<512, 256, 0, stream>>>(Wih, Whh, bih, bhh, WihB, WhhB, bias);
    prep_state  <<<512, 256, 0, stream>>>(h0a, c0a, h0b, c0b, Hb, C);
    prep_embed  <<<2048, 256, 0, stream>>>(s1, s2, emb, X);

    for (int t = 0; t < T_STEPS; ++t)
        lstm_step<<<dim3(16, 4), 256, 0, stream>>>(WihB, WhhB, bias, X, Hb, C,
                                                   Hf, t);

    final_out<<<BATCH, 256, 0, stream>>>(Hf, out);
}
